// SGMViTHybridModel_6451040878709
// MI455X (gfx1250) — matrix operand order repaired and checked by reading, not compiled, not run on hardware
//
#include <hip/hip_runtime.h>
#include <hip/hip_bf16.h>
#include <math.h>

// ---------------- model constants ----------------
#define BB 4
#define GG 28
#define PPATCH 14
#define NN 784          // tokens per image (28*28)
#define BN 3136         // BB*NN
#define DD 768
#define HH 12
#define HDIM 64
#define TD 2304         // 3*DD
#define MLPD 3072
#define LL 4
#define KPATCH 588      // 3*14*14
#define KPADP 608       // padded to multiple of 32
#define IMGS 392
#define NHEADPAD 256    // head out 196 padded to 256
#define LDY 208
#define QT 49           // 784/16 key/query tiles

typedef __attribute__((ext_vector_type(16))) __bf16 v16bf;
typedef __attribute__((ext_vector_type(8)))  __bf16 v8bf;
typedef __attribute__((ext_vector_type(8)))  float  v8f;
typedef __attribute__((ext_vector_type(4)))  int    i32x4;

#define WMMA_BF16(A_, B_, C_) \
  __builtin_amdgcn_wmma_f32_16x16x32_bf16(false, (A_), false, (B_), (short)0, (C_), false, false)

// ---- CDNA5 async global->LDS path (guarded; falls back to sync staging) ----
#if defined(__HIP_DEVICE_COMPILE__) && __has_builtin(__builtin_amdgcn_global_load_async_to_lds_b128)
#define HAS_ASYNC_LDS 1
#else
#define HAS_ASYNC_LDS 0
#endif

__device__ inline void async_copy_b128(const __bf16* gsrc, __bf16* lds_dst) {
#if HAS_ASYNC_LDS
  __builtin_amdgcn_global_load_async_to_lds_b128(
      (__attribute__((address_space(1))) i32x4*)gsrc,
      (__attribute__((address_space(3))) i32x4*)lds_dst,
      0, 0);
#else
  *(v8bf*)lds_dst = *(const v8bf*)gsrc;
#endif
}

__device__ inline void wait_async_lds() {
#if HAS_ASYNC_LDS
#if __has_builtin(__builtin_amdgcn_s_wait_asynccnt)
  __builtin_amdgcn_s_wait_asynccnt(0);
#else
  asm volatile("s_wait_asynccnt 0" ::: "memory");
#endif
#endif
}

// ---------------- fragment loaders ----------------
// A fragment 16x32 bf16 (ISA 7.12.2): lane row = lane&15, half = lane>>4,
// VGPRs 0-3 hold K = 8*half + [0..7], VGPRs 4-7 hold K = 16 + 8*half + [0..7].
__device__ inline v16bf load_a_frag(const __bf16* base, int lda, int m0, int kbase, int lane) {
  int row  = m0 + (lane & 15);
  int half = lane >> 4;
  const __bf16* p = base + (size_t)row * lda + kbase + 8 * half;
  v8bf lo = *(const v8bf*)p;          // K run [8h, 8h+8)
  v8bf hi = *(const v8bf*)(p + 16);   // K run [16+8h, 16+8h+8)
  return __builtin_shufflevector(lo, hi, 0,1,2,3,4,5,6,7,8,9,10,11,12,13,14,15);
}

__device__ inline float gelu_tanh(float x) {
  float x3 = x * x * x;
  return 0.5f * x * (1.0f + tanhf(0.7978845608028654f * (x + 0.044715f * x3)));
}

// ---------------- weight convert + transpose: W(out,in) f32 -> Wt(inPad,outPad) bf16 ----------------
__global__ void k_w2bf(__bf16* __restrict__ dst, const float* __restrict__ src,
                       int OUT, int IN, int NPAD, int KP) {
  int idx = blockIdx.x * 256 + threadIdx.x;
  if (idx >= KP * NPAD) return;
  int k = idx / NPAD, n = idx % NPAD;
  float v = (k < IN && n < OUT) ? src[(size_t)n * IN + k] : 0.0f;
  dst[idx] = (__bf16)v;
}

// ---------------- im2col for patch embed: A(BN x KPADP) bf16 ----------------
__global__ void k_im2col(const float* __restrict__ img, __bf16* __restrict__ A) {
  int idx = blockIdx.x * 256 + threadIdx.x;   // BN*KPADP exact
  int k = idx % KPADP;
  int t = idx / KPADP;
  float v = 0.0f;
  if (k < KPATCH) {
    int b = t / NN, n = t % NN;
    int gy = n / GG, gx = n % GG;
    int c = k / 196, r = k % 196;
    int py = r / PPATCH, px = r % PPATCH;
    v = img[(((size_t)b * 3 + c) * IMGS + gy * PPATCH + py) * IMGS + gx * PPATCH + px];
  }
  A[idx] = (__bf16)v;
}

// ---------------- router ----------------
__global__ void k_zero_i32(int* p, int n) {
  int i = blockIdx.x * 256 + threadIdx.x;
  if (i < n) p[i] = 0;
}

__global__ void k_router(const float* __restrict__ conf, float* __restrict__ conf_out,
                         int* __restrict__ keep, int* __restrict__ cnt, int* __restrict__ anyk) {
  int idx = blockIdx.x * 256 + threadIdx.x;
  if (idx >= BN) return;
  int b = idx / NN, n = idx % NN;
  int gy = n / GG, gx = n % GG;
  const float* base = conf + ((size_t)b * IMGS + gy * PPATCH) * IMGS + gx * PPATCH;
  float s = 0.0f;
  for (int py = 0; py < PPATCH; ++py)
    for (int px = 0; px < PPATCH; ++px) s += base[py * IMGS + px];
  float mean = s * (1.0f / 196.0f);
  conf_out[idx] = mean;
  int kp = (mean >= 0.6f) ? 1 : 0;
  keep[idx] = kp;
  if (kp) { atomicAdd(cnt, 1); atomicOr(&anyk[b], 1); }
}

__global__ void k_prune(const int* cnt, float* out_scalar) {
  if (threadIdx.x == 0) *out_scalar = 1.0f - (float)(*cnt) / (float)BN;
}

// ---------------- layernorm f32 -> bf16 ----------------
__global__ __launch_bounds__(256) void k_ln(const float* __restrict__ x,
                                            const float* __restrict__ sc,
                                            const float* __restrict__ bi,
                                            __bf16* __restrict__ y) {
  __shared__ float r1[256], r2[256];
  int tid = threadIdx.x;
  const float* xr = x + (size_t)blockIdx.x * DD;
  float a0 = xr[tid], a1 = xr[tid + 256], a2 = xr[tid + 512];
  r1[tid] = a0 + a1 + a2;
  r2[tid] = a0 * a0 + a1 * a1 + a2 * a2;
  __syncthreads();
  for (int off = 128; off > 0; off >>= 1) {
    if (tid < off) { r1[tid] += r1[tid + off]; r2[tid] += r2[tid + off]; }
    __syncthreads();
  }
  float mean = r1[0] * (1.0f / DD);
  float var  = r2[0] * (1.0f / DD) - mean * mean;
  float rs = rsqrtf(var + 1e-6f);
  __bf16* yr = y + (size_t)blockIdx.x * DD;
  yr[tid]       = (__bf16)((a0 - mean) * rs * sc[tid]       + bi[tid]);
  yr[tid + 256] = (__bf16)((a1 - mean) * rs * sc[tid + 256] + bi[tid + 256]);
  yr[tid + 512] = (__bf16)((a2 - mean) * rs * sc[tid + 512] + bi[tid + 512]);
}

// ---------------- WMMA bf16 GEMM: C(M x N) = A(M x K) * Wt(K x N) + bias [+res] [gelu] ----------------
// grid: (N/256, 49). block tile 64(M) x 256(N); 8 waves, each 64x32 -> 8 accumulators.
// A tile (64x32, shared by all 8 waves) is staged through LDS with double-buffered
// async global->LDS copies (ASYNCcnt); B streams straight from L2 per wave.
__global__ __launch_bounds__(256) void k_gemm(const __bf16* __restrict__ A, int lda,
                                              const __bf16* __restrict__ Wt, int ldb,
                                              const float* __restrict__ bias,
                                              const float* __restrict__ res, int ldres,
                                              float* __restrict__ Cf, __bf16* __restrict__ Cb, int ldc,
                                              int K, int trueN, int act) {
  __shared__ __align__(16) __bf16 Asm[2][64 * 32];
  int tid  = threadIdx.x;
  int lane = tid & 31;
  int wid  = tid >> 5;
  int n_wave = blockIdx.x * 256 + wid * 32;
  int m0 = blockIdx.y * 64;

  // staging: 256 threads x 16B = the whole 4KB A tile
  int crow = tid >> 2;                       // 0..63
  int cseg = tid & 3;                        // 0..3 (8 bf16 each)
  const __bf16* asrc = A + (size_t)(m0 + crow) * lda + cseg * 8;
  __bf16* adst[2] = { &Asm[0][crow * 32 + cseg * 8], &Asm[1][crow * 32 + cseg * 8] };

  v8f acc[8];
#pragma unroll
  for (int i = 0; i < 8; ++i) acc[i] = {};

  // per-lane B pointer (lane&15 = N column, 16 K rows gathered in table order)
  const __bf16* bp = Wt + n_wave + (lane & 15);
  const size_t bstep = (size_t)32 * ldb;

  async_copy_b128(asrc, adst[0]);            // prime k = 0

  int half = lane >> 4;
  int arow = lane & 15;

  int cb = 0;
  for (int k = 0; k < K; k += 32, cb ^= 1) {
    wait_async_lds();
    __syncthreads();                         // tile cb ready; everyone done with cb^1
    int knext = (k + 32 < K) ? (k + 32) : k; // clamped (last iter copy is dead but safe)
    async_copy_b128(asrc + knext, adst[cb ^ 1]);

    v16bf b0, b1;
#pragma unroll
    for (int s = 0; s < 16; ++s) {
      int kk = (s < 8) ? (8 * half + s) : (8 + 8 * half + s);
      b0[s] = bp[(size_t)kk * ldb];
      b1[s] = bp[(size_t)kk * ldb + 16];
    }
    bp += bstep;

    const __bf16* at = &Asm[cb][0];
#pragma unroll
    for (int i = 0; i < 4; ++i) {
      const __bf16* ap = at + (16 * i + arow) * 32 + 8 * half;
      v8bf lo = *(const v8bf*)ap;
      v8bf hi = *(const v8bf*)(ap + 16);
      v16bf af = __builtin_shufflevector(lo, hi, 0,1,2,3,4,5,6,7,8,9,10,11,12,13,14,15);
      acc[2 * i]     = WMMA_BF16(af, b0, acc[2 * i]);
      acc[2 * i + 1] = WMMA_BF16(af, b1, acc[2 * i + 1]);
    }
  }

  int coll = lane & 15;
#pragma unroll
  for (int i = 0; i < 4; ++i) {
#pragma unroll
    for (int j = 0; j < 2; ++j) {
      int n = n_wave + 16 * j + coll;
      if (n < trueN) {
        float bv = bias[n];
#pragma unroll
        for (int e = 0; e < 8; ++e) {
          int m = m0 + 16 * i + 8 * half + e;   // C layout: VGPR e -> row e + 8*half
          float v = acc[i * 2 + j][e] + bv;
          if (act == 1) v = gelu_tanh(v);
          if (res) v += res[(size_t)m * ldres + n];
          if (Cf) Cf[(size_t)m * ldc + n] = v;
          else    Cb[(size_t)m * ldc + n] = (__bf16)v;
        }
      }
    }
  }
}

// ---------------- K transpose: qkv(b,n, 768 + h*64 + d) -> Kt[((b*12+h)*64+d)*784 + n] ----------------
__global__ void k_ktrans(const __bf16* __restrict__ qkv, __bf16* __restrict__ Kt) {
  int idx = blockIdx.x * 256 + threadIdx.x;   // BB*HH*HDIM*NN exact
  int n = idx % NN;
  int r = idx / NN;
  int d = r % HDIM;
  int bh = r / HDIM;
  int h = bh % HH, b = bh / HH;
  Kt[idx] = qkv[(size_t)(b * NN + n) * TD + DD + h * HDIM + d];
}

// ---------------- flash attention: one wave = 16 queries of one (b,h) ----------------
__global__ __launch_bounds__(256) void k_flash(const __bf16* __restrict__ qkv,
                                               const __bf16* __restrict__ Kt,
                                               const int* __restrict__ keep,
                                               __bf16* __restrict__ outb, int use_mask) {
  __shared__ float plds[8][16][16];
  int lane = threadIdx.x & 31;
  int wid  = threadIdx.x >> 5;
  int gw = blockIdx.x * 8 + wid;            // 294 blocks * 8 = 2352 = BB*HH*QT exact
  int qt = gw % QT;
  int h  = (gw / QT) % HH;
  int b  = gw / (QT * HH);
  int q0 = qt * 16;
  int half = lane >> 4;
  int coll = lane & 15;

  const __bf16* qbase = qkv + (size_t)b * NN * TD + h * HDIM;
  v16bf qa0 = load_a_frag(qbase, TD, q0, 0, lane);
  v16bf qa1 = load_a_frag(qbase, TD, q0, 32, lane);

  const __bf16* ktb   = Kt + (size_t)(b * HH + h) * HDIM * NN;
  const __bf16* vbase = qkv + (size_t)b * NN * TD + 2 * DD + h * HDIM;

  float mrow[8], lrow[8];
  v8f o[4];
#pragma unroll
  for (int e = 0; e < 8; ++e) { mrow[e] = -3.0e38f; lrow[e] = 0.0f; }
#pragma unroll
  for (int t = 0; t < 4; ++t) o[t] = {};

  for (int nc = 0; nc < QT; ++nc) {
    int n0 = nc * 16;
    v8f s = {};
    v16bf kb0, kb1;
#pragma unroll
    for (int s = 0; s < 16; ++s) {
      int kk = (s < 8) ? (8 * half + s) : (8 + 8 * half + s);
      kb0[s] = ktb[(size_t)kk * NN + n0 + coll];
      kb1[s] = ktb[(size_t)(32 + kk) * NN + n0 + coll];
    }
    s = WMMA_BF16(qa0, kb0, s);
    s = WMMA_BF16(qa1, kb1, s);

    int key = n0 + coll;
    bool masked = use_mask && (keep[b * NN + key] == 0);
#pragma unroll
    for (int e = 0; e < 8; ++e) {
      float v = s[e] * 0.125f;                 // 1/sqrt(64)
      if (masked) v = -1.0e9f;
      float rm = v;
      rm = fmaxf(rm, __shfl_xor(rm, 8, 16));
      rm = fmaxf(rm, __shfl_xor(rm, 4, 16));
      rm = fmaxf(rm, __shfl_xor(rm, 2, 16));
      rm = fmaxf(rm, __shfl_xor(rm, 1, 16));
      float mn  = fmaxf(mrow[e], rm);
      float esc = expf(mrow[e] - mn);
      float p   = expf(v - mn);
      float rs = p;
      rs += __shfl_xor(rs, 8, 16);
      rs += __shfl_xor(rs, 4, 16);
      rs += __shfl_xor(rs, 2, 16);
      rs += __shfl_xor(rs, 1, 16);
      lrow[e] = lrow[e] * esc + rs;
      mrow[e] = mn;
#pragma unroll
      for (int t = 0; t < 4; ++t) o[t][e] *= esc;
      plds[wid][8 * half + e][coll] = p;       // C layout -> [query][key]
    }
    __builtin_amdgcn_wave_barrier();
    asm volatile("s_wait_dscnt 0" ::: "memory");
    // rebuild P as A fragment (16 queries x 32 keys; keys 16..31 = 0)
    v16bf pa;
#pragma unroll
    for (int i = 0; i < 8; ++i) {
      pa[i]     = (__bf16)plds[wid][coll][8 * half + i];
      pa[8 + i] = (__bf16)0.0f;
    }
    __builtin_amdgcn_wave_barrier();
#pragma unroll
    for (int t = 0; t < 4; ++t) {
      v16bf vb;
#pragma unroll
      for (int s = 0; s < 16; ++s) {
        int krow = n0 + ((s < 8) ? (8 * half + s) : (8 + 8 * half + s));
        if (krow > NN - 1) krow = NN - 1;        // P is zero there anyway
        vb[s] = vbase[(size_t)krow * TD + t * 16 + coll];
      }
      o[t] = WMMA_BF16(pa, vb, o[t]);
    }
  }

#pragma unroll
  for (int t = 0; t < 4; ++t) {
#pragma unroll
    for (int e = 0; e < 8; ++e) {
      int row = q0 + 8 * half + e;
      float v = o[t][e] / lrow[e];
      outb[(size_t)(b * NN + row) * DD + h * HDIM + t * 16 + coll] = (__bf16)v;
    }
  }
}

// ---------------- sparse select after block 0 ----------------
__global__ void k_select(const float* __restrict__ xA, const float* __restrict__ tokens,
                         const float* __restrict__ fill, const int* __restrict__ keep,
                         const int* __restrict__ anyk, float* __restrict__ x) {
  int idx = blockIdx.x * 256 + threadIdx.x;   // BN*DD exact
  int t = idx / DD, d = idx % DD;
  int b = t / NN;
  float v;
  if (!anyk[b]) v = tokens[idx];
  else v = keep[t] ? xA[idx] : fill[d];
  x[idx] = v;
}

// ---------------- reshape head output to depth map ----------------
__global__ void k_reshape(const float* __restrict__ y, float* __restrict__ out) {
  int idx = blockIdx.x * 256 + threadIdx.x;   // BB*IMGS*IMGS exact
  int c = idx % IMGS;
  int r = (idx / IMGS) % IMGS;
  int b = idx / (IMGS * IMGS);
  int gy = r / PPATCH, py = r % PPATCH;
  int gx = c / PPATCH, px = c % PPATCH;
  int tok = b * NN + gy * GG + gx;
  out[idx] = y[(size_t)tok * LDY + py * PPATCH + px];
}

// ================= host =================
extern "C" void kernel_launch(void* const* d_in, const int* in_sizes, int n_in,
                              void* d_out, int out_size, void* d_ws, size_t ws_size,
                              hipStream_t stream) {
  (void)in_sizes; (void)n_in; (void)out_size; (void)ws_size;

  const float* image   = (const float*)d_in[0];
  const float* confmap = (const float*)d_in[1];
  const float* patch_w = (const float*)d_in[2];
  const float* patch_b = (const float*)d_in[3];
  const float* ln1_s   = (const float*)d_in[4];
  const float* ln1_b   = (const float*)d_in[5];
  const float* qkv_w   = (const float*)d_in[6];
  const float* qkv_b   = (const float*)d_in[7];
  const float* proj_w  = (const float*)d_in[8];
  const float* proj_b  = (const float*)d_in[9];
  const float* ln2_s   = (const float*)d_in[10];
  const float* ln2_b   = (const float*)d_in[11];
  const float* fc1_w   = (const float*)d_in[12];
  const float* fc1_b   = (const float*)d_in[13];
  const float* fc2_w   = (const float*)d_in[14];
  const float* fc2_b   = (const float*)d_in[15];
  const float* norm_s  = (const float*)d_in[16];
  const float* norm_b  = (const float*)d_in[17];
  const float* fillt   = (const float*)d_in[18];
  const float* head_w  = (const float*)d_in[19];
  const float* head_b  = (const float*)d_in[20];
  float* out = (float*)d_out;

  char* ws = (char*)d_ws;
  size_t cur = 0;
  auto alloc = [&](size_t bytes) -> char* {
    cur = (cur + 255) & ~(size_t)255;
    char* p = ws + cur;
    cur += bytes;
    return p;
  };

  __bf16* wQKV  = (__bf16*)alloc((size_t)LL * DD * TD * 2);
  __bf16* wPROJ = (__bf16*)alloc((size_t)LL * DD * DD * 2);
  __bf16* wFC1  = (__bf16*)alloc((size_t)LL * DD * MLPD * 2);
  __bf16* wFC2  = (__bf16*)alloc((size_t)LL * MLPD * DD * 2);
  __bf16* wHEAD = (__bf16*)alloc((size_t)DD * NHEADPAD * 2);
  __bf16* wPAT  = (__bf16*)alloc((size_t)KPADP * DD * 2);
  __bf16* Apat  = (__bf16*)alloc((size_t)BN * KPADP * 2);
  float*  tokens= (float*)alloc((size_t)BN * DD * 4);
  float*  x     = (float*)alloc((size_t)BN * DD * 4);
  float*  xA    = (float*)alloc((size_t)BN * DD * 4);
  float*  xT    = (float*)alloc((size_t)BN * DD * 4);
  __bf16* xln   = (__bf16*)alloc((size_t)BN * DD * 2);
  __bf16* qkvbuf= (__bf16*)alloc((size_t)BN * TD * 2);
  __bf16* Kt    = (__bf16*)alloc((size_t)BB * HH * HDIM * NN * 2);
  __bf16* attnb = (__bf16*)alloc((size_t)BN * DD * 2);
  __bf16* h1    = (__bf16*)alloc((size_t)BN * MLPD * 2);
  float*  ybuf  = (float*)alloc((size_t)BN * LDY * 4);
  int*    keep  = (int*)alloc((size_t)BN * 4);
  int*    cnts  = (int*)alloc(32);            // [0]=keep count, [1..4]=any_keep per batch

  float* out_depth = out;                               // 4*392*392
  float* out_prune = out + (size_t)BB * IMGS * IMGS;    // 1
  float* out_conf  = out_prune + 1;                     // 4*28*28

  // ---- weight conversion (transpose to K-major bf16) ----
  for (int l = 0; l < LL; ++l) {
    k_w2bf<<<(DD * TD) / 256, 256, 0, stream>>>(wQKV + (size_t)l * DD * TD,
        qkv_w + (size_t)l * TD * DD, TD, DD, TD, DD);
    k_w2bf<<<(DD * DD) / 256, 256, 0, stream>>>(wPROJ + (size_t)l * DD * DD,
        proj_w + (size_t)l * DD * DD, DD, DD, DD, DD);
    k_w2bf<<<(DD * MLPD) / 256, 256, 0, stream>>>(wFC1 + (size_t)l * DD * MLPD,
        fc1_w + (size_t)l * MLPD * DD, MLPD, DD, MLPD, DD);
    k_w2bf<<<(MLPD * DD) / 256, 256, 0, stream>>>(wFC2 + (size_t)l * MLPD * DD,
        fc2_w + (size_t)l * DD * MLPD, DD, MLPD, DD, MLPD);
  }
  k_w2bf<<<(DD * NHEADPAD) / 256, 256, 0, stream>>>(wHEAD, head_w, 196, DD, NHEADPAD, DD);
  k_w2bf<<<(KPADP * DD) / 256, 256, 0, stream>>>(wPAT, patch_w, DD, KPATCH, DD, KPADP);

  // ---- im2col + router ----
  k_im2col<<<(BN * KPADP) / 256, 256, 0, stream>>>(image, Apat);
  k_zero_i32<<<1, 256, 0, stream>>>(cnts, 8);
  k_router<<<(BN + 255) / 256, 256, 0, stream>>>(confmap, out_conf, keep, &cnts[0], &cnts[1]);
  k_prune<<<1, 32, 0, stream>>>(&cnts[0], out_prune);

  // ---- patch embed GEMM -> tokens ----
  {
    dim3 g(DD / 256, BN / 64);
    k_gemm<<<g, 256, 0, stream>>>(Apat, KPADP, wPAT, DD, patch_b,
                                  nullptr, 0, tokens, nullptr, DD, KPADP, DD, 0);
  }

  // ---- transformer block runner ----
  auto run_block = [&](int l, const float* xin, float* xout, int use_mask) {
    k_ln<<<BN, 256, 0, stream>>>(xin, ln1_s + l * DD, ln1_b + l * DD, xln);
    {
      dim3 g(TD / 256, BN / 64);
      k_gemm<<<g, 256, 0, stream>>>(xln, DD, wQKV + (size_t)l * DD * TD, TD,
                                    qkv_b + (size_t)l * TD, nullptr, 0,
                                    nullptr, qkvbuf, TD, DD, TD, 0);
    }
    k_ktrans<<<(BB * HH * HDIM * NN) / 256, 256, 0, stream>>>(qkvbuf, Kt);
    k_flash<<<(BB * HH * QT) / 8, 256, 0, stream>>>(qkvbuf, Kt, keep, attnb, use_mask);
    {
      dim3 g(DD / 256, BN / 64);
      k_gemm<<<g, 256, 0, stream>>>(attnb, DD, wPROJ + (size_t)l * DD * DD, DD,
                                    proj_b + (size_t)l * DD, xin, DD,
                                    xT, nullptr, DD, DD, DD, 0);
    }
    k_ln<<<BN, 256, 0, stream>>>(xT, ln2_s + l * DD, ln2_b + l * DD, xln);
    {
      dim3 g(MLPD / 256, BN / 64);
      k_gemm<<<g, 256, 0, stream>>>(xln, DD, wFC1 + (size_t)l * DD * MLPD, MLPD,
                                    fc1_b + (size_t)l * MLPD, nullptr, 0,
                                    nullptr, h1, MLPD, DD, MLPD, 1);
    }
    {
      dim3 g(DD / 256, BN / 64);
      k_gemm<<<g, 256, 0, stream>>>(h1, MLPD, wFC2 + (size_t)l * MLPD * DD, DD,
                                    fc2_b + (size_t)l * DD, xT, DD,
                                    xout, nullptr, DD, MLPD, DD, 0);
    }
  };

  // block 0 with key mask on tokens -> xA, then sparse select -> x
  run_block(0, tokens, xA, 1);
  k_select<<<(BN * DD) / 256, 256, 0, stream>>>(xA, tokens, fillt, keep, &cnts[1], x);

  // full encoder pass (block 0 runs again, as in reference)
  for (int l = 0; l < LL; ++l) run_block(l, x, x, 0);

  // final LN + head GEMM + reshape
  k_ln<<<BN, 256, 0, stream>>>(x, norm_s, norm_b, xln);
  {
    dim3 g(NHEADPAD / 256, BN / 64);
    k_gemm<<<g, 256, 0, stream>>>(xln, DD, wHEAD, NHEADPAD, head_b,
                                  nullptr, 0, ybuf, nullptr, LDY, DD, 196, 0);
  }
  k_reshape<<<(BB * IMGS * IMGS) / 256, 256, 0, stream>>>(ybuf, out_depth);
}
